// CgpHmmLayer_68341519614148
// MI455X (gfx1250) — matrix operand order repaired and checked by reading, not compiled, not run on hardware
//
#include <hip/hip_runtime.h>

// ---------------- constants ----------------
#define BATCH   128
#define T_LEN   4096
#define S_DIM   308
#define SP      320           // padded states
#define EM_DIM  6
#define NT      20            // N tiles of 16
#define KT      10            // K iterations of 32
#define AST     328           // alpha LDS row stride (halfwords) -> conflict-free b128 loads

typedef __attribute__((ext_vector_type(16))) __bf16 v16bf;
typedef __attribute__((ext_vector_type(8)))  float  v8f;

union Frag { uint4 q[2]; v16bf v; };

__device__ __forceinline__ unsigned short f2bf(float f) {
    unsigned int u = __float_as_uint(f);
    unsigned int r = (u + 0x7FFFu + ((u >> 16) & 1u)) >> 16;   // round-to-nearest-even
    return (unsigned short)r;
}

// ---------------- setup kernels ----------------

// One block (512 threads) per row; softmax along the row.
__global__ void softmax_rows(const float* __restrict__ in, float* __restrict__ out, int cols) {
    int r = blockIdx.x, tid = threadIdx.x;
    __shared__ float red[512];
    float x = (tid < cols) ? in[r * cols + tid] : -3.4e38f;
    red[tid] = x; __syncthreads();
    for (int s = 256; s > 0; s >>= 1) { if (tid < s) red[tid] = fmaxf(red[tid], red[tid + s]); __syncthreads(); }
    float mx = red[0]; __syncthreads();
    float e = (tid < cols) ? __expf(x - mx) : 0.0f;
    red[tid] = e; __syncthreads();
    for (int s = 256; s > 0; s >>= 1) { if (tid < s) red[tid] += red[tid + s]; __syncthreads(); }
    float sum = red[0];
    if (tid < cols) out[r * cols + tid] = e / sum;
}

// Etab[e*SP + n] = Bsoft[n*EM + e], zero-padded for n >= S_DIM.
__global__ void build_etab(const float* __restrict__ Bsoft, float* __restrict__ Etab) {
    int i = blockIdx.x * blockDim.x + threadIdx.x;
    if (i < EM_DIM * SP) {
        int e = i / SP, n = i % SP;
        Etab[i] = (n < S_DIM) ? Bsoft[n * EM_DIM + e] : 0.0f;
    }
}

// Pack A into per-fragment WMMA-B layout (bf16). Fragment f = kt*NT + nt.
// Lane l, halfword j: n = nt*16 + (l&15); k = kt*32 + (j&7) + (j>>3)*16 + (l<16 ? 0 : 8).
__global__ void pack_A(const float* __restrict__ Asoft, unsigned short* __restrict__ Apk) {
    int f = blockIdx.x;              // 0..199
    int kt = f / NT, nt = f % NT;
    int tid = threadIdx.x;           // 0..511
    int l = tid >> 4, j = tid & 15;
    int n = nt * 16 + (l & 15);
    int k = kt * 32 + (j & 7) + ((j >> 3) << 4) + ((l < 16) ? 0 : 8);
    float v = (k < S_DIM && n < S_DIM) ? Asoft[k * S_DIM + n] : 0.0f;
    Apk[f * 512 + tid] = f2bf(v);
}

// obsT[t*BATCH + b] = obs[b*T + t]
__global__ void transpose_obs(const int* __restrict__ obs, int* __restrict__ obsT) {
    int i = blockIdx.x * blockDim.x + threadIdx.x;
    if (i < BATCH * T_LEN) {
        int t = i / BATCH, b = i % BATCH;
        obsT[i] = obs[b * T_LEN + t];
    }
}

// ---------------- main forward kernel ----------------
// Grid: 8 workgroups x 320 threads (10 waves). Each WG owns 16 batch rows and
// runs the full T-step recursion. alpha in LDS (bf16), A streamed from L2 in
// pre-packed WMMA-B fragment order, all matmuls via v_wmma_f32_16x16x32_bf16.
__global__ __launch_bounds__(320) void hmm_forward(const int* __restrict__ obsT,
                                                   const float* __restrict__ Isoft,
                                                   const float* __restrict__ Etab_g,
                                                   const uint4* __restrict__ Apk,
                                                   float* __restrict__ out) {
    __shared__ __align__(16) unsigned short s_alpha[16 * AST];
    __shared__ float s_E[EM_DIM * SP];
    __shared__ float s_rs[2][16];
    __shared__ float s_ll[16];
    __shared__ int   s_sym[16];

    const int tid  = threadIdx.x;
    const int lane = tid & 31;
    const int w    = tid >> 5;              // wave id 0..9
    const int wgbase = blockIdx.x * 16;

    for (int i = tid; i < EM_DIM * SP; i += 320) s_E[i] = Etab_g[i];
    if (tid < 16) {
        s_sym[tid] = obsT[wgbase + tid];    // t = 0 symbols
        s_ll[tid] = 0.0f;
        s_rs[0][tid] = 0.0f;
        s_rs[1][tid] = 0.0f;
    }
    __syncthreads();

    // ---- t = 0 : alpha = I * E(o0), scaled ----
    {
        float vals[16];
        int n = tid;                         // 0..319 (one column per thread)
        float iv = (n < S_DIM) ? Isoft[n] : 0.0f;
        #pragma unroll
        for (int m = 0; m < 16; ++m) {
            float v = iv * s_E[s_sym[m] * SP + n];
            vals[m] = v;
            atomicAdd(&s_rs[0][m], v);
        }
        __syncthreads();
        if (tid < 16) s_ll[tid] = __logf(s_rs[0][tid]);
        #pragma unroll
        for (int m = 0; m < 16; ++m) {
            float z = s_rs[0][m];
            s_alpha[m * AST + n] = f2bf(vals[m] / z);
        }
        __syncthreads();
        if (tid < 16) s_rs[0][tid] = 0.0f;
    }

    // per-thread geometry
    const int nt0 = w * 2, nt1 = w * 2 + 1;
    const int mrow = lane & 15;                 // a-frag row
    const int koff = (lane >> 4) << 3;          // 0 or 8
    const int nn0 = nt0 * 16 + (lane & 15);
    const int nn1 = nt1 * 16 + (lane & 15);
    const int mo  = (lane >> 4) * 8;            // C row offset per lane half

    for (int t = 1; t < T_LEN; ++t) {
        const int p = t & 1;
        if (tid < 16) s_sym[tid] = obsT[t * BATCH + wgbase + tid];
        __syncthreads();                        // sym visible; prev alpha/rs settled

        v8f acc0, acc1;
        #pragma unroll
        for (int i = 0; i < 8; ++i) { acc0[i] = 0.0f; acc1[i] = 0.0f; }

        #pragma unroll
        for (int kt = 0; kt < KT; ++kt) {
            const int Kb = kt * 32;
            Frag a, b0, b1;
            a.q[0] = *(const uint4*)(&s_alpha[mrow * AST + Kb + koff]);
            a.q[1] = *(const uint4*)(&s_alpha[mrow * AST + Kb + 16 + koff]);
            const uint4* pb0 = Apk + (((kt * NT + nt0) * 32 + lane) << 1);
            const uint4* pb1 = Apk + (((kt * NT + nt1) * 32 + lane) << 1);
            b0.q[0] = pb0[0]; b0.q[1] = pb0[1];
            b1.q[0] = pb1[0]; b1.q[1] = pb1[1];
            acc0 = __builtin_amdgcn_wmma_f32_16x16x32_bf16(false, a.v, false, b0.v,
                                                           (short)0, acc0, false, false);
            acc1 = __builtin_amdgcn_wmma_f32_16x16x32_bf16(false, a.v, false, b1.v,
                                                           (short)0, acc1, false, false);
        }

        // ---- emission scale + row sums ----
        float v0[8], v1[8], ps[8];
        #pragma unroll
        for (int r = 0; r < 8; ++r) {
            int m = r + mo;
            int sy = s_sym[m];
            v0[r] = acc0[r] * s_E[sy * SP + nn0];
            v1[r] = acc1[r] * s_E[sy * SP + nn1];
            float s = v0[r] + v1[r];
            s += __shfl_xor(s, 1, 32);
            s += __shfl_xor(s, 2, 32);
            s += __shfl_xor(s, 4, 32);
            s += __shfl_xor(s, 8, 32);          // full sum within each 16-lane half
            ps[r] = s;
        }
        if ((lane & 15) == 0) {
            #pragma unroll
            for (int r = 0; r < 8; ++r) atomicAdd(&s_rs[p][r + mo], ps[r]);
        }
        __syncthreads();                        // row sums complete

        if (tid < 16) s_ll[tid] += __logf(s_rs[p][tid]);
        #pragma unroll
        for (int r = 0; r < 8; ++r) {
            int m = r + mo;
            float inv = 1.0f / s_rs[p][m];
            s_alpha[m * AST + nn0] = f2bf(v0[r] * inv);
            s_alpha[m * AST + nn1] = f2bf(v1[r] * inv);
        }
        __syncthreads();                        // alpha rewritten
        if (tid < 16) s_rs[p][tid] = 0.0f;      // safe: next use of rs[p] is t+2
    }

    __syncthreads();
    if (tid < 16) out[wgbase + tid] = s_ll[tid];
}

// ---------------- launcher ----------------
extern "C" void kernel_launch(void* const* d_in, const int* in_sizes, int n_in,
                              void* d_out, int out_size, void* d_ws, size_t ws_size,
                              hipStream_t stream) {
    const int*   obs      = (const int*)d_in[0];
    const float* I_kernel = (const float*)d_in[1];
    const float* A_kernel = (const float*)d_in[2];
    const float* B_kernel = (const float*)d_in[3];
    float* out = (float*)d_out;

    char* w = (char*)d_ws;
    float*          Isoft = (float*)(w);                       // 308 f (reserve 1280 B)
    float*          Etab  = (float*)(w + 1280);                // 6*320 f = 7680 B
    unsigned short* Apk   = (unsigned short*)(w + 8960);       // 200*1024 B = 204800 B
    float*          Asoft = (float*)(w + 213760);              // 308*308 f = 379456 B
    float*          Bsoft = (float*)(w + 593216);              // 308*6 f   = 7392 B (rsv 7424)
    int*            obsT  = (int*)(w + 600640);                // 4096*128 i = 2097152 B

    softmax_rows<<<1,   512, 0, stream>>>(I_kernel, Isoft, S_DIM);
    softmax_rows<<<S_DIM, 512, 0, stream>>>(A_kernel, Asoft, S_DIM);
    softmax_rows<<<S_DIM, 512, 0, stream>>>(B_kernel, Bsoft, EM_DIM);
    build_etab<<<(EM_DIM * SP + 255) / 256, 256, 0, stream>>>(Bsoft, Etab);
    pack_A<<<KT * NT, 512, 0, stream>>>(Asoft, Apk);
    transpose_obs<<<(BATCH * T_LEN + 255) / 256, 256, 0, stream>>>(obs, obsT);

    hmm_forward<<<BATCH / 16, 320, 0, stream>>>(obsT, Isoft, Etab, (const uint4*)Apk, out);
}
